// FeedForward_88493506167135
// MI455X (gfx1250) — matrix operand order repaired and checked by reading, not compiled, not run on hardware
//
#include <hip/hip_runtime.h>
#include <math.h>

#define M_TILE  32
#define THREADS 256

// Padded LDS row strides (elements). bf16 strides: dwords ≡ 4 (mod 8) -> the 16
// rows of a WMMA A-fragment hit 16 distinct 4-bank groups (64-bank LDS), and each
// row stays 16B-aligned so ds_load_b128 is legal. f32 stride 652: 12*r mod 64
// distinct for r=0..15.
#define LDF 648   // feat   (logical 640)
#define LDH 1544  // h      (logical 1536)
#define LDG 1288  // gated  (logical 1280)
#define LDO 652   // out    (logical 640)

typedef __bf16 bf16_t;
typedef bf16_t v16bf __attribute__((ext_vector_type(16)));
typedef float  v8f   __attribute__((ext_vector_type(8)));

struct __align__(16) U128 { unsigned int a, b, c, d; };
union __align__(16) FragCvt { U128 u[2]; v16bf v; };

__device__ __forceinline__ v8f wmma_bf16(v8f acc, v16bf a, v16bf b) {
  // (neg_a, A, neg_b, B, c_mod, C, reuse_a, reuse_b)
  return __builtin_amdgcn_wmma_f32_16x16x32_bf16(false, a, false, b, (short)0, acc, false, false);
}

// Two 16x16 output tiles (rows 0..15 and 16..31) sharing the same B fragments.
// A row-major bf16 in LDS; B pre-transposed (BT[n][k], row stride K) bf16 in global (L2-resident).
// A frag (16-bit A 16x32): lane l -> row (l&15), K = koff..koff+7 and koff+16..koff+23, koff=(l>>4)*8.
// B frag (16-bit B 32x16): lane l -> col (l&15), K = kb..kb+7 and kb+16..kb+23, kb=(l>>4)*8.
__device__ __forceinline__ void tile_gemm_pair(const bf16_t* __restrict__ A, int lda,
                                               const bf16_t* __restrict__ BT, int K, int col0,
                                               v8f& acc0, v8f& acc1) {
  const int lane = threadIdx.x & 31;
  const bf16_t* pa = A + (lane & 15) * lda + ((lane >> 4) * 8);
  const bf16_t* pb = BT + (size_t)(col0 + (lane & 15)) * K + ((lane >> 4) * 8);
  const int ldh = 16 * lda;
  for (int k0 = 0; k0 < K; k0 += 32) {
    FragCvt fb, fa0, fa1;
    fb.u[0]  = *(const U128*)(pb + k0);
    fb.u[1]  = *(const U128*)(pb + k0 + 16);
    fa0.u[0] = *(const U128*)(pa + k0);
    fa0.u[1] = *(const U128*)(pa + k0 + 16);
    fa1.u[0] = *(const U128*)(pa + ldh + k0);
    fa1.u[1] = *(const U128*)(pa + ldh + k0 + 16);
    acc0 = wmma_bf16(acc0, fa0.v, fb.v);
    acc1 = wmma_bf16(acc1, fa1.v, fb.v);
  }
}

// f32 16x16 C/D layout: lane l -> col (l&15); VGPR j -> row (l>>4)*8 + j.
__device__ __forceinline__ void store_tile_bf16(bf16_t* D, int ldd, int row0, int col0, v8f acc) {
  const int lane = threadIdx.x & 31;
  const int c = lane & 15, r0 = (lane >> 4) * 8;
  #pragma unroll
  for (int j = 0; j < 8; ++j)
    D[(row0 + r0 + j) * ldd + col0 + c] = (bf16_t)acc[j];
}

__device__ __forceinline__ void store_tile_f32(float* D, int ldd, int row0, int col0, v8f acc) {
  const int lane = threadIdx.x & 31;
  const int c = lane & 15, r0 = (lane >> 4) * 8;
  #pragma unroll
  for (int j = 0; j < 8; ++j)
    D[(row0 + r0 + j) * ldd + col0 + c] = acc[j];
}

// gate: sG[r][0..511] = silu(h[0..511]); sG planes d: v_d * sigmoid(h[512..767])
__device__ __forceinline__ void gate_apply(const bf16_t* sH, bf16_t* sG, int tid) {
  for (int e = tid; e < M_TILE * 1280; e += THREADS) {
    int r = e / 1280, c = e - r * 1280;
    const bf16_t* hr = sH + r * LDH;
    float o;
    if (c < 512) {
      float x = (float)hr[c];
      o = x / (1.f + __expf(-x));
    } else {
      int q = c - 512; int d = q >> 8; int i = q & 255;
      float g = 1.f / (1.f + __expf(-(float)hr[512 + i]));
      o = (float)hr[768 + (d << 8) + i] * g;
    }
    sG[r * LDG + c] = (bf16_t)o;
  }
}

__global__ void zero_stats_k(float* stats) {
  int i = blockIdx.x * blockDim.x + threadIdx.x;
  if (i < 1280) stats[i] = 0.f;
}

// W (K x N, f32 row-major) -> WT (N x K, bf16 row-major), scale folded in
__global__ void prep_wt_k(const float* __restrict__ W, bf16_t* __restrict__ WT,
                          int K, int N, float scale) {
  int idx = blockIdx.x * blockDim.x + threadIdx.x;
  if (idx < K * N) {
    int n = idx / K;
    int k = idx - n * K;
    WT[idx] = (bf16_t)(W[(size_t)k * N + n] * scale);
  }
}

__global__ void __launch_bounds__(THREADS)
ffn_main_k(const float* __restrict__ features, float* __restrict__ out,
           float* __restrict__ stats, int n,
           const bf16_t* __restrict__ w1a, const bf16_t* __restrict__ w1b,
           const bf16_t* __restrict__ w2a, const bf16_t* __restrict__ w2b,
           const bf16_t* __restrict__ w3a, const bf16_t* __restrict__ w3b,
           const bf16_t* __restrict__ wba, const bf16_t* __restrict__ wbb) {
  extern __shared__ char smem[];
  bf16_t* sFeat = (bf16_t*)smem;                // 32 x LDF  (s256 | d0 128 | d1 128 | d2 128)
  bf16_t* sH    = sFeat + M_TILE * LDF;         // 32 x LDH  (s768 | d0 256 | d1 256 | d2 256)
  bf16_t* sG    = sH + M_TILE * LDH;            // 32 x LDG  (s512 | d0 256 | d1 256 | d2 256)
  float*  sOut  = (float*)(sG + M_TILE * LDG);  // 32 x LDO  (s256 | d0 128 | d1 128 | d2 128)

  const int tid = threadIdx.x;
  const int wave = tid >> 5;
  const int rowBase = blockIdx.x * M_TILE;

  // ---- load features, de-interleave vector channels into planes, fp32->bf16
  for (int e = tid; e < M_TILE * 640; e += THREADS) {
    int r = e / 640, c = e - r * 640;
    int gr = rowBase + r;
    float x = 0.f;
    if (gr < n) {
      int src = (c < 256) ? c : 256 + ((c - 256) & 127) * 3 + ((c - 256) >> 7);
      x = features[(size_t)gr * 640 + src];
    }
    sFeat[r * LDF + c] = (bf16_t)x;
  }
  __syncthreads();

  // ---- GEMM1: h1 = lin1(feat)  (scales folded into WT). 96 column tiles, 2 row tiles each
  for (int c = wave; c < 96; c += 8) {
    v8f a0 = {}, a1 = {};
    if (c < 48) {                   // scalar: feat_s(32x256) @ l1W0T(768x256)
      tile_gemm_pair(sFeat, LDF, w1a, 256, c * 16, a0, a1);
      store_tile_bf16(sH, LDH, 0, c * 16, a0);
      store_tile_bf16(sH, LDH, 16, c * 16, a1);
    } else {                        // vector plane d: feat_d(32x128) @ l1W1T(256x128)
      int d = (c - 48) >> 4, cc = (c - 48) & 15;
      tile_gemm_pair(sFeat + 256 + 128 * d, LDF, w1b, 128, cc * 16, a0, a1);
      store_tile_bf16(sH, LDH, 0, 768 + 256 * d + cc * 16, a0);
      store_tile_bf16(sH, LDH, 16, 768 + 256 * d + cc * 16, a1);
    }
  }
  __syncthreads();

  gate_apply(sH, sG, tid);
  __syncthreads();

  // ---- GEMM2: h2 = lin2(g1). 96 column tiles
  for (int c = wave; c < 96; c += 8) {
    v8f a0 = {}, a1 = {};
    if (c < 48) {                   // g1_s(32x512) @ l2W0T(768x512)
      tile_gemm_pair(sG, LDG, w2a, 512, c * 16, a0, a1);
      store_tile_bf16(sH, LDH, 0, c * 16, a0);
      store_tile_bf16(sH, LDH, 16, c * 16, a1);
    } else {                        // g1_d(32x256) @ l2W1T(256x256)
      int d = (c - 48) >> 4, cc = (c - 48) & 15;
      tile_gemm_pair(sG + 512 + 256 * d, LDG, w2b, 256, cc * 16, a0, a1);
      store_tile_bf16(sH, LDH, 0, 768 + 256 * d + cc * 16, a0);
      store_tile_bf16(sH, LDH, 16, 768 + 256 * d + cc * 16, a1);
    }
  }
  __syncthreads();

  gate_apply(sH, sG, tid);
  __syncthreads();

  // ---- GEMM3 + bypass fused into one accumulator per tile. 40 column tiles
  for (int c = wave; c < 40; c += 8) {
    v8f a0 = {}, a1 = {};
    if (c < 16) {                   // scalar out: g2_s @ l3W0T + feat_s @ bypW0T
      tile_gemm_pair(sG, LDG, w3a, 512, c * 16, a0, a1);
      tile_gemm_pair(sFeat, LDF, wba, 256, c * 16, a0, a1);
      store_tile_f32(sOut, LDO, 0, c * 16, a0);
      store_tile_f32(sOut, LDO, 16, c * 16, a1);
    } else {                        // vector plane d: g2_d @ l3W1T + feat_d @ bypW1T
      int d = (c - 16) >> 3, cc = (c - 16) & 7;
      tile_gemm_pair(sG + 512 + 256 * d, LDG, w3b, 256, cc * 16, a0, a1);
      tile_gemm_pair(sFeat + 256 + 128 * d, LDF, wbb, 128, cc * 16, a0, a1);
      store_tile_f32(sOut, LDO, 0, 256 + 128 * d + cc * 16, a0);
      store_tile_f32(sOut, LDO, 16, 256 + 128 * d + cc * 16, a1);
    }
  }
  __syncthreads();

  // ---- write pre-batchnorm out (re-interleaved) + accumulate global stats
  for (int e = tid; e < M_TILE * 640; e += THREADS) {
    int r = e / 640, c = e - r * 640;
    int gr = rowBase + r;
    if (gr < n) {
      int dst = (c < 256) ? c : 256 + ((c - 256) & 127) * 3 + ((c - 256) >> 7);
      out[(size_t)gr * 640 + dst] = sOut[r * LDO + c];
    }
  }
  {
    // scalar channels: sum and sumsq (OOB rows are exactly zero -> neutral)
    float s = 0.f, sq = 0.f;
    for (int r = 0; r < M_TILE; ++r) {
      float x = sOut[r * LDO + tid];
      s += x; sq += x * x;
    }
    atomicAdd(&stats[tid], s);
    atomicAdd(&stats[256 + tid], sq);
    if (tid < 128) {
      float vq = 0.f;
      for (int r = 0; r < M_TILE; ++r)
        #pragma unroll
        for (int d = 0; d < 3; ++d) {
          float x = sOut[r * LDO + 256 + (d << 7) + tid];
          vq += x * x;
        }
      atomicAdd(&stats[512 + tid], vq);
    }
  }
}

__global__ void finalize_k(const float* __restrict__ stats, float* __restrict__ fin,
                           const float* __restrict__ bnw, float invN) {
  int t = threadIdx.x;
  if (t < 256) {
    float m = stats[t] * invN;
    float var = stats[256 + t] * invN - m * m;
    fin[t] = m;                                   // mean_s
    fin[256 + t] = rsqrtf(var + 1e-5f) * bnw[t];  // s_inv * w
  } else if (t < 384) {
    int ch = t - 256;
    float ms = stats[512 + ch] * invN * (1.f / 3.f);
    fin[512 + ch] = rsqrtf(ms + 1e-5f) * bnw[256 + ch];  // v_inv * w
  }
}

__global__ void apply_bn_k(float* __restrict__ out, const float* __restrict__ fin,
                           const float* __restrict__ bnb, size_t total) {
  size_t idx = (size_t)blockIdx.x * blockDim.x + threadIdx.x;
  if (idx >= total) return;
  int c = (int)(idx % 640);
  float x = out[idx];
  if (c < 256) out[idx] = (x - fin[c]) * fin[256 + c] + bnb[c];
  else {
    int i = (c - 256) / 3;
    out[idx] = x * fin[512 + i];
  }
}

extern "C" void kernel_launch(void* const* d_in, const int* in_sizes, int n_in,
                              void* d_out, int out_size, void* d_ws, size_t ws_size,
                              hipStream_t stream) {
  (void)n_in; (void)out_size; (void)ws_size;
  const float* features = (const float*)d_in[0];
  const float* bnw = (const float*)d_in[9];
  const float* bnb = (const float*)d_in[10];
  float* out = (float*)d_out;
  const int n = in_sizes[0] / 640;

  float* stats = (float*)d_ws;        // 640 f32: sum_s[256] | sumsq_s[256] | sumsq_v[128]
  float* fin = stats + 640;           // 640 f32: mean_s[256] | sinv_w[256] | vinv_w[128]
  bf16_t* wt = (bf16_t*)((char*)d_ws + 8192);

  // pre-transposed, scale-folded bf16 weights (N x K each)
  bf16_t* w1a = wt + 0;        // lin1_W0T 768x256
  bf16_t* w1b = wt + 196608;   // lin1_W1T 256x128
  bf16_t* w2a = wt + 229376;   // lin2_W0T 768x512
  bf16_t* w2b = wt + 622592;   // lin2_W1T 256x256
  bf16_t* w3a = wt + 688128;   // lin3_W0T 256x512
  bf16_t* w3b = wt + 819200;   // lin3_W1T 128x256
  bf16_t* wba = wt + 851968;   // byp_W0T  256x256
  bf16_t* wbb = wt + 917504;   // byp_W1T  128x128

  zero_stats_k<<<(1280 + 255) / 256, 256, 0, stream>>>(stats);

  struct Cfg { int K, N; float scale; bf16_t* dst; const float* src; };
  const float s128 = 1.f / sqrtf(128.f), s256 = 1.f / 16.f, s512 = 1.f / sqrtf(512.f);
  Cfg cfg[8] = {
    {256, 768, s256, w1a, (const float*)d_in[1]},
    {128, 256, s128, w1b, (const float*)d_in[2]},
    {512, 768, s512, w2a, (const float*)d_in[3]},
    {256, 256, s256, w2b, (const float*)d_in[4]},
    {512, 256, s512, w3a, (const float*)d_in[5]},
    {256, 128, s256, w3b, (const float*)d_in[6]},
    {256, 256, s256, wba, (const float*)d_in[7]},
    {128, 128, s128, wbb, (const float*)d_in[8]},
  };
  for (int i = 0; i < 8; ++i) {
    int total = cfg[i].K * cfg[i].N;
    prep_wt_k<<<(total + 255) / 256, 256, 0, stream>>>(cfg[i].src, cfg[i].dst,
                                                       cfg[i].K, cfg[i].N, cfg[i].scale);
  }

  const size_t smem = (size_t)M_TILE * LDF * 2 + (size_t)M_TILE * LDH * 2 +
                      (size_t)M_TILE * LDG * 2 + (size_t)M_TILE * LDO * 4;  // 306176 B
  const int blocks = (n + M_TILE - 1) / M_TILE;
  ffn_main_k<<<blocks, THREADS, smem, stream>>>(features, out, stats, n,
                                                w1a, w1b, w2a, w2b, w3a, w3b, wba, wbb);

  finalize_k<<<1, 384, 0, stream>>>(stats, fin, bnw, 1.f / (float)n);

  const size_t total = (size_t)n * 640;
  apply_bn_k<<<(unsigned)((total + 255) / 256), 256, 0, stream>>>(out, fin, bnb, total);
}
